// BrainAgeGNN_12678743458068
// MI455X (gfx1250) — matrix operand order repaired and checked by reading, not compiled, not run on hardware
//
#include <hip/hip_runtime.h>

typedef __attribute__((ext_vector_type(16))) _Float16 v16h;
typedef __attribute__((ext_vector_type(8)))  _Float16 v8h;
typedef __attribute__((ext_vector_type(8)))  float    v8f;

#define EPS 1e-5f

// ---------------- elementwise / scatter kernels ----------------

__global__ void fill_f32(float* __restrict__ p, float v, int n) {
    int t = blockIdx.x * blockDim.x + threadIdx.x;
    if (t < n) p[t] = v;
}

// deg[dst] += w   (deg pre-filled with 1.0 for the self-loop)
__global__ void deg_scatter(const int* __restrict__ dst, const float* __restrict__ w,
                            float* __restrict__ deg, int E) {
    int t = blockIdx.x * blockDim.x + threadIdx.x;
    if (t < E) atomicAdd(&deg[dst[t]], w[t]);
}

// in-place deg -> dinv
__global__ void deg_to_dinv(float* __restrict__ deg, int N) {
    int t = blockIdx.x * blockDim.x + threadIdx.x;
    if (t < N) deg[t] = rsqrtf(deg[t]);
}

__global__ void edge_norm(const int* __restrict__ src, const int* __restrict__ dst,
                          const float* __restrict__ w, const float* __restrict__ dinv,
                          float* __restrict__ nrm, int E) {
    int t = blockIdx.x * blockDim.x + threadIdx.x;
    if (t < E) nrm[t] = dinv[src[t]] * w[t] * dinv[dst[t]];
}

// layer-1 linear: lin[n,c] = x[n] * W1[c]   (rank-1, F=64)
__global__ void layer1_lin(const float* __restrict__ x, const float* __restrict__ W1,
                           float* __restrict__ lin, int N) {
    int t = blockIdx.x * blockDim.x + threadIdx.x;
    if (t >= (N << 6)) return;
    int n = t >> 6, c = t & 63;
    lin[(size_t)n * 64 + c] = x[n] * W1[c];
}

// agg[dst,F] += lin[src,F] * norm[e]; one thread = 4 features of one edge
__global__ void agg_scatter(const int* __restrict__ src, const int* __restrict__ dst,
                            const float* __restrict__ nrm, const float* __restrict__ lin,
                            float* __restrict__ agg, int E, int cshift /*4:F=64, 5:F=128*/) {
    int t = blockIdx.x * blockDim.x + threadIdx.x;
    int total = E << cshift;
    if (t >= total) return;
    int e = t >> cshift;
    int c = (t & ((1 << cshift) - 1)) << 2;
    int F = 4 << cshift;
    int s = src[e], d = dst[e];
    float nm = nrm[e];
    const float4 h4 = *(const float4*)(lin + (size_t)s * F + c);
    float* ap = agg + (size_t)d * F + c;
    atomicAdd(ap + 0, h4.x * nm);
    atomicAdd(ap + 1, h4.y * nm);
    atomicAdd(ap + 2, h4.z * nm);
    atomicAdd(ap + 3, h4.w * nm);
}

// epilogue: val = relu(bn(agg + lin*dinv^2 + b)) [+ res]; writes f32 and/or f16
__global__ void post_bn_relu(const float* __restrict__ lin, const float* __restrict__ agg,
                             const float* __restrict__ dinv,
                             const float* __restrict__ b, const float* __restrict__ g,
                             const float* __restrict__ be, const float* __restrict__ mu,
                             const float* __restrict__ var,
                             const float* __restrict__ res,
                             float* __restrict__ out32, _Float16* __restrict__ out16,
                             int N, int fshift) {
    int t = blockIdx.x * blockDim.x + threadIdx.x;
    if (t >= (N << fshift)) return;
    int n = t >> fshift;
    int c = t & ((1 << fshift) - 1);
    size_t idx = (size_t)t;
    float di = dinv[n];
    float val = agg[idx] + lin[idx] * di * di + b[c];
    val = (val - mu[c]) * rsqrtf(var[c] + EPS) * g[c] + be[c];
    val = fmaxf(val, 0.0f);
    if (res)  val += res[idx];
    if (out32) out32[idx] = val;
    if (out16) out16[idx] = (_Float16)val;
}

// pack W[K,Fout] (row-major f32) into WMMA B fragments:
// Bp[kcIdx][tn][lane][e] = (f16) W[kc*32 + (e/8)*16 + (lane/16)*8 + e%8][tn*16 + lane%16]
__global__ void pack_b(const float* __restrict__ W, _Float16* __restrict__ Bp,
                       int K, int Fout) {
    int t = blockIdx.x * blockDim.x + threadIdx.x;
    if (t >= K * Fout) return;
    int e = t & 15;
    int lane = (t >> 4) & 31;
    int rest = t >> 9;                 // kcIdx * nTilesN + tn
    int nTilesN = Fout >> 4;
    int tn = rest % nTilesN;
    int kcIdx = rest / nTilesN;
    int k = (kcIdx << 5) + ((e >> 3) << 4) + ((lane >> 4) << 3) + (e & 7);
    int n = (tn << 4) + (lane & 15);
    Bp[t] = (_Float16)W[(size_t)k * Fout + n];
}

// ---------------- WMMA GEMM: C[N,Fout] = A[N,K](f16) x B(packed f16), f32 accum ----------------
// block = 128 threads (4 waves); wave w handles row tile blockIdx.x*4+w, blockIdx.y = col tile.
__global__ void gemm_wmma_f16(const _Float16* __restrict__ A, const _Float16* __restrict__ Bp,
                              float* __restrict__ C, int N, int K, int Fout) {
    const int lane = threadIdx.x & 31;
    const int wave = threadIdx.x >> 5;
    const int rowTile = blockIdx.x * 4 + wave;
    const int nRowTiles = N >> 4;
    if (rowTile >= nRowTiles) return;          // uniform per-wave exit: EXEC stays all-ones
    const int colTile = blockIdx.y;
    const int half = lane >> 4;
    const int l15 = lane & 15;
    const int nTilesN = Fout >> 4;
    const _Float16* arow = A + (size_t)((rowTile << 4) + l15) * K;

    v8f acc = {};
    for (int kc = 0; kc < K; kc += 32) {
        union { v16h v; v8h h[2]; } a, b;
        // A fragment: e<8 -> K = kc + half*8 + e ; e>=8 -> K = kc + 16 + half*8 + (e-8)
        a.h[0] = *(const v8h*)(arow + kc + half * 8);
        a.h[1] = *(const v8h*)(arow + kc + 16 + half * 8);
        // B fragment: contiguous 32B per lane from the packed layout
        const _Float16* bptr = Bp + (((size_t)(kc >> 5) * nTilesN + colTile) * 32 + lane) * 16;
        b.h[0] = *(const v8h*)(bptr);
        b.h[1] = *(const v8h*)(bptr + 8);
        acc = __builtin_amdgcn_wmma_f32_16x16x32_f16(
            false, a.v, false, b.v, (short)0, acc, false, false);
    }
    // D layout: VGPR r -> row (rowTile*16 + half*8 + r), col (colTile*16 + l15)
    const int nBase = (colTile << 4) + l15;
#pragma unroll
    for (int r = 0; r < 8; ++r) {
        int mo = (rowTile << 4) + half * 8 + r;
        C[(size_t)mo * Fout + nBase] = acc[r];
    }
}

// ---------------- pooling + head ----------------

__global__ void pool_scatter(const float* __restrict__ h, const int* __restrict__ batch,
                             float* __restrict__ pooled, int N) {
    int t = blockIdx.x * blockDim.x + threadIdx.x;
    if (t >= (N << 7)) return;
    int n = t >> 7, c = t & 127;
    atomicAdd(&pooled[(size_t)batch[n] * 128 + c], h[(size_t)t]);
}

__global__ void pool_count(const int* __restrict__ batch, float* __restrict__ cnt, int N) {
    int t = blockIdx.x * blockDim.x + threadIdx.x;
    if (t < N) atomicAdd(&cnt[batch[t]], 1.0f);
}

__global__ void head_linear(const float* __restrict__ pooled, const float* __restrict__ cnt,
                            const float* __restrict__ Wf, const float* __restrict__ bf,
                            float* __restrict__ out, int G) {
    int g = blockIdx.x * blockDim.x + threadIdx.x;
    if (g >= G) return;
    float inv = 1.0f / fmaxf(cnt[g], 1.0f);
    float acc = 0.0f;
    const float* p = pooled + (size_t)g * 128;
#pragma unroll 4
    for (int c = 0; c < 128; ++c) acc += p[c] * inv * Wf[c];
    out[g] = acc + bf[0];
}

// ---------------- launch ----------------

extern "C" void kernel_launch(void* const* d_in, const int* in_sizes, int n_in,
                              void* d_out, int out_size, void* d_ws, size_t ws_size,
                              hipStream_t stream) {
    const float* x    = (const float*)d_in[0];
    const int*   ei   = (const int*)  d_in[1];
    const float* ew   = (const float*)d_in[2];
    const int*   bat  = (const int*)  d_in[3];
    const float* W1   = (const float*)d_in[4];
    const float* b1   = (const float*)d_in[5];
    const float* W2   = (const float*)d_in[6];
    const float* b2   = (const float*)d_in[7];
    const float* W3   = (const float*)d_in[8];
    const float* b3   = (const float*)d_in[9];
    const float* Wf   = (const float*)d_in[10];
    const float* bf   = (const float*)d_in[11];
    const float* g1   = (const float*)d_in[12];
    const float* be1  = (const float*)d_in[13];
    const float* m1   = (const float*)d_in[14];
    const float* v1   = (const float*)d_in[15];
    const float* g2   = (const float*)d_in[16];
    const float* be2  = (const float*)d_in[17];
    const float* m2   = (const float*)d_in[18];
    const float* v2   = (const float*)d_in[19];
    const float* g3   = (const float*)d_in[20];
    const float* be3  = (const float*)d_in[21];
    const float* m3   = (const float*)d_in[22];
    const float* v3   = (const float*)d_in[23];

    const int N = in_sizes[0];
    const int E = in_sizes[2];
    const int G = out_size;                 // output [G, 1] float32
    const int* src = ei;
    const int* dst = ei + E;

    // workspace carve (256B aligned slabs)
    char* ws = (char*)d_ws;
    size_t off = 0;
    auto carve = [&](size_t bytes) -> void* {
        void* p = ws + off;
        off += (bytes + 255) & ~(size_t)255;
        return p;
    };
    float*     deg    = (float*)    carve((size_t)N * 4);          // becomes dinv in place
    float*     nrm    = (float*)    carve((size_t)E * 4);
    float*     lin    = (float*)    carve((size_t)N * 128 * 4);
    float*     agg    = (float*)    carve((size_t)N * 128 * 4);
    _Float16*  act16  = (_Float16*) carve((size_t)N * 128 * 2);
    float*     res32  = (float*)    carve((size_t)N * 128 * 4);
    float*     fin    = (float*)    carve((size_t)N * 128 * 4);
    _Float16*  W2p    = (_Float16*) carve((size_t)64 * 128 * 2);
    _Float16*  W3p    = (_Float16*) carve((size_t)128 * 128 * 2);
    float*     pooled = (float*)    carve((size_t)G * 128 * 4);
    float*     cnt    = (float*)    carve((size_t)G * 4);
    (void)ws_size;

    auto blk = [](long long total, int t) { return (unsigned)((total + t - 1) / t); };
    const int T = 256;

    // --- degree / norm ---
    fill_f32<<<blk(N, T), T, 0, stream>>>(deg, 1.0f, N);            // self-loop weight
    deg_scatter<<<blk(E, T), T, 0, stream>>>(dst, ew, deg, E);
    deg_to_dinv<<<blk(N, T), T, 0, stream>>>(deg, N);               // deg -> dinv
    edge_norm<<<blk(E, T), T, 0, stream>>>(src, dst, ew, deg, nrm, E);

    // --- layer 1 (F=64): lin = x @ W1 ; aggregate ; BN+ReLU -> act16 ---
    layer1_lin<<<blk((long long)N * 64, T), T, 0, stream>>>(x, W1, lin, N);
    fill_f32<<<blk((long long)N * 64, T), T, 0, stream>>>(agg, 0.0f, N * 64);
    agg_scatter<<<blk((long long)E * 16, T), T, 0, stream>>>(src, dst, nrm, lin, agg, E, 4);
    post_bn_relu<<<blk((long long)N * 64, T), T, 0, stream>>>(
        lin, agg, deg, b1, g1, be1, m1, v1, nullptr, nullptr, act16, N, 6);

    // --- layer 2 (64 -> 128): WMMA GEMM ; aggregate ; BN+ReLU -> res32 + act16 ---
    pack_b<<<blk(64 * 128, T), T, 0, stream>>>(W2, W2p, 64, 128);
    {
        int rowTiles = N >> 4;
        dim3 grid((rowTiles + 3) / 4, 128 / 16);
        gemm_wmma_f16<<<grid, 128, 0, stream>>>(act16, W2p, lin, N, 64, 128);
    }
    fill_f32<<<blk((long long)N * 128, T), T, 0, stream>>>(agg, 0.0f, N * 128);
    agg_scatter<<<blk((long long)E * 32, T), T, 0, stream>>>(src, dst, nrm, lin, agg, E, 5);
    post_bn_relu<<<blk((long long)N * 128, T), T, 0, stream>>>(
        lin, agg, deg, b2, g2, be2, m2, v2, nullptr, res32, act16, N, 7);

    // --- layer 3 (128 -> 128): WMMA GEMM ; aggregate ; BN+ReLU + residual -> fin ---
    pack_b<<<blk(128 * 128, T), T, 0, stream>>>(W3, W3p, 128, 128);
    {
        int rowTiles = N >> 4;
        dim3 grid((rowTiles + 3) / 4, 128 / 16);
        gemm_wmma_f16<<<grid, 128, 0, stream>>>(act16, W3p, lin, N, 128, 128);
    }
    fill_f32<<<blk((long long)N * 128, T), T, 0, stream>>>(agg, 0.0f, N * 128);
    agg_scatter<<<blk((long long)E * 32, T), T, 0, stream>>>(src, dst, nrm, lin, agg, E, 5);
    post_bn_relu<<<blk((long long)N * 128, T), T, 0, stream>>>(
        lin, agg, deg, b3, g3, be3, m3, v3, res32, fin, nullptr, N, 7);

    // --- global mean pool + head ---
    fill_f32<<<blk((long long)G * 128, T), T, 0, stream>>>(pooled, 0.0f, G * 128);
    fill_f32<<<blk(G, T), T, 0, stream>>>(cnt, 0.0f, G);
    pool_scatter<<<blk((long long)N * 128, T), T, 0, stream>>>(fin, bat, pooled, N);
    pool_count<<<blk(N, T), T, 0, stream>>>(bat, cnt, N);
    head_linear<<<blk(G, T), T, 0, stream>>>(pooled, cnt, Wf, bf, (float*)d_out, G);
}
